// ScaleSpaceDetector_47485158424873
// MI455X (gfx1250) — matrix operand order repaired and checked by reading, not compiled, not run on hardware
//
#include <hip/hip_runtime.h>
#include <hip/hip_bf16.h>
#include <math.h>

// ---------------------------------------------------------------------------
// CDNA5 (gfx1250) scale-space Hessian detector.
// Gaussian separable blur implemented as banded-matrix GEMM via
// v_wmma_f32_16x16x32_f16 (wave32, 16x16 tiles, f32 accumulate).
// Fragment data paths arranged as aligned 16B LDS vector loads (ds_load_b128).
// Hessian tile staging uses gfx1250 async global->LDS loads (ASYNCcnt).
// ---------------------------------------------------------------------------

typedef _Float16 v16h __attribute__((ext_vector_type(16)));
typedef _Float16 v8h  __attribute__((ext_vector_type(8)));
typedef float    v8f  __attribute__((ext_vector_type(8)));

#define BTILE 32                 // workgroup output tile (32x32)
#define PMAX  12                 // max blur radius (ksize<=25)
#define INMAX (BTILE + 2*PMAX)   // 56 rows max staged
#define SINW  80                 // sIn LDS stride (f16): covers col reads <=79
#define SMW   96                 // sMid LDS stride (f16): covers row reads <=79

static __device__ __forceinline__ int iclamp(int v, int lo, int hi) {
    return v < lo ? lo : (v > hi ? hi : v);
}

#define SHUF16(lo, hi) __builtin_shufflevector((lo), (hi), 0, 1, 2, 3, 4, 5, \
                                               6, 7, 8, 9, 10, 11, 12, 13, 14, 15)

// ---------------------------------------------------------------------------
// Fused separable Gaussian blur. One workgroup = 4 waves = 32x32 output tile.
// Stage 1: stage (32+2p) x 80 input region into LDS as f16 (clamp padding).
// Stage 2: horizontal pass: per 16x16 tile, D = In(16x32k) x Gband(32k x16),
//          one v_wmma_f32_16x16x32_f16 per 32-wide K chunk; store transposed.
// Stage 3: vertical pass: D = Gband(16x32k) x Mid(32k x16); store f32.
// ---------------------------------------------------------------------------
__global__ __launch_bounds__(128) void k_blur(
    const float* __restrict__ src, float* __restrict__ dst,
    int H, int W, long long srcStride, long long dstStride, float sigma)
{
    __shared__ __align__(16) _Float16 sIn[INMAX * SINW];
    __shared__ __align__(16) _Float16 sMid[BTILE * SMW];
    __shared__ float sG[32];

    const int tid = threadIdx.x;
    int ks = (int)(8.0f * sigma + 1.0f);
    if ((ks & 1) == 0) ks++;
    if (ks > 25) ks = 25;
    const int p = ks >> 1;

    // pre-zero sMid (stage-3 reads rows beyond the written region with zero
    // band coefficients; operands must be finite)
    {
        unsigned long long* mz = (unsigned long long*)sMid;
        for (int i = tid; i < (BTILE * SMW) / 4; i += 128) mz[i] = 0ull;
    }
    if (tid < ks) {
        float inv2s2 = 0.5f / (sigma * sigma);
        float s = 0.f;
        for (int k = 0; k < ks; ++k) {
            float x = (float)(k - p);
            s += __expf(-x * x * inv2s2);
        }
        float x = (float)(tid - p);
        sG[tid] = __expf(-x * x * inv2s2) / s;
    }

    const float* ims = src + (long long)blockIdx.z * srcStride;
    float*       imd = dst + (long long)blockIdx.z * dstStride;
    const int y0 = blockIdx.y * BTILE - p;
    const int x0 = blockIdx.x * BTILE - p;
    const int inR = BTILE + 2 * p;
    const int inC = inR;
    if (tid == 0) __builtin_prefetch(ims + (long long)iclamp(y0, 0, H - 1) * W, 0, 0);

    // Stage 1: global -> LDS (f16), full 80-wide rows so fragment loads never
    // need clamping (cols >= inC replicate the edge; all finite)
    for (int i = tid; i < inR * SINW; i += 128) {
        int r = i / SINW, c = i - r * SINW;
        int cc = c < inC ? c : inC - 1;
        int gr = iclamp(r + y0, 0, H - 1);
        int gc = iclamp(cc + x0, 0, W - 1);
        sIn[i] = (_Float16)ims[(long long)gr * W + gc];
    }
    __syncthreads();

    const int wid  = tid >> 5;
    const int lane = tid & 31;
    const int lrow = lane & 15;
    const int lhi  = lane >> 4;
    const int nrt  = (inR + 15) >> 4;          // row tiles for H pass (<=4)
    const int nkc  = (16 + 2 * p + 31) >> 5;   // K chunks of 32 (<=2)
    const int kba  = lhi ? 8 : 0;              // A-fragment K base (16-bit A)
    const int kbb  = lhi ? 16 : 0;             // stage-3 K base for A and B (contiguous k = 16h + s)

    // Band-matrix B fragments for the horizontal pass (per K chunk).
    // B[idx] holds K = kba + idx + (idx>=8 ? 8:0); coefficient g(kc*32 + K - n), n = lrow.
    v16h hb0, hb1;
    #pragma unroll
    for (int idx = 0; idx < 16; ++idx) {
        int gi0 = kba + idx + (idx >= 8 ? 8 : 0) - lrow;
        int gi1 = 32 + kba + idx + (idx >= 8 ? 8 : 0) - lrow;
        hb0[idx] = (_Float16)((gi0 >= 0 && gi0 < ks) ? sG[gi0] : 0.f);
        hb1[idx] = (_Float16)((gi1 >= 0 && gi1 < ks) ? sG[gi1] : 0.f);
    }
    // Band-matrix A fragments for the vertical pass (per K chunk).
    // A[idx] holds K = kbb + idx; coeff g(kc*32 + K - m).
    v16h va0, va1;
    #pragma unroll
    for (int idx = 0; idx < 16; ++idx) {
        int K = kbb + idx;
        int gi0 = K - lrow;
        int gi1 = 32 + K - lrow;
        va0[idx] = (_Float16)((gi0 >= 0 && gi0 < ks) ? sG[gi0] : 0.f);
        va1[idx] = (_Float16)((gi1 >= 0 && gi1 < ks) ? sG[gi1] : 0.f);
    }

    // Stage 2: horizontal pass -> sMid (stored transposed [col][row])
    for (int t = wid; t < nrt * 2; t += 4) {
        const int rt = t >> 1, ct = t & 1;
        int arow = rt * 16 + lrow;
        if (arow > inR - 1) arow = inR - 1;
        const _Float16* rowp = &sIn[arow * SINW];
        v8f acc = {};
        {
            const int cb = ct * 16 + kba;                  // 16B aligned
            v8h lo = *(const v8h*)&rowp[cb];
            v8h hi = *(const v8h*)&rowp[cb + 16];
            v16h a = SHUF16(lo, hi);
            acc = __builtin_amdgcn_wmma_f32_16x16x32_f16(
                false, a, false, hb0, (short)0, acc, false, false);
        }
        if (nkc > 1) {
            const int cb = ct * 16 + 32 + kba;             // max read col 79
            v8h lo = *(const v8h*)&rowp[cb];
            v8h hi = *(const v8h*)&rowp[cb + 16];
            v16h a = SHUF16(lo, hi);
            acc = __builtin_amdgcn_wmma_f32_16x16x32_f16(
                false, a, false, hb1, (short)0, acc, false, false);
        }
        v8h acch;
        #pragma unroll
        for (int i = 0; i < 8; ++i) acch[i] = (_Float16)acc[i];
        // rows rt*16 + 8*lhi + (0..7) are consecutive in the transposed layout
        *(v8h*)&sMid[(ct * 16 + lrow) * SMW + rt * 16 + 8 * lhi] = acch;
    }
    __syncthreads();

    // Stage 3: vertical pass (band matrix is A, mid data is B)
    {
        const int rt = wid >> 1, ct = wid & 1;
        const _Float16* colp = &sMid[(ct * 16 + lrow) * SMW];
        v8f acc = {};
        {
            const int rb = rt * 16 + kbb;                  // 16B aligned
            v8h lo = *(const v8h*)&colp[rb];
            v8h hi = *(const v8h*)&colp[rb + 8];
            v16h b = SHUF16(lo, hi);
            acc = __builtin_amdgcn_wmma_f32_16x16x32_f16(
                false, va0, false, b, (short)0, acc, false, false);
        }
        if (nkc > 1) {
            const int rb = rt * 16 + 32 + kbb;             // max read row 79
            v8h lo = *(const v8h*)&colp[rb];
            v8h hi = *(const v8h*)&colp[rb + 8];
            v16h b = SHUF16(lo, hi);
            acc = __builtin_amdgcn_wmma_f32_16x16x32_f16(
                false, va1, false, b, (short)0, acc, false, false);
        }
        const int col = blockIdx.x * BTILE + ct * 16 + lrow;
        #pragma unroll
        for (int i = 0; i < 8; ++i) {
            int row = blockIdx.y * BTILE + rt * 16 + i + (lhi ? 8 : 0);
            if (row < H && col < W) imd[(long long)row * W + col] = acc[i];
        }
    }
}

// ---------------------------------------------------------------------------
// Hessian determinant response (5x5 stencil, clamp/edge padding) * sigma^4.
// Tile staged through LDS; uses gfx1250 async global->LDS loads if available.
// ---------------------------------------------------------------------------
static __device__ const float GXXc[5][5] = {
    {-0.015625f, 0.f, 0.03125f, 0.f, -0.015625f},
    {-0.0625f,   0.f, 0.125f,   0.f, -0.0625f},
    {-0.09375f,  0.f, 0.1875f,  0.f, -0.09375f},
    {-0.0625f,   0.f, 0.125f,   0.f, -0.0625f},
    {-0.015625f, 0.f, 0.03125f, 0.f, -0.015625f}};
static __device__ const float GXYc[5][5] = {
    {-0.02777778f, -0.05555556f, 0.f,  0.05555556f,  0.02777778f},
    {-0.05555556f, -0.11111111f, 0.f,  0.11111111f,  0.05555556f},
    { 0.f,          0.f,         0.f,  0.f,          0.f},
    { 0.05555556f,  0.11111111f, 0.f, -0.11111111f, -0.05555556f},
    { 0.02777778f,  0.05555556f, 0.f, -0.05555556f, -0.02777778f}};
static __device__ const float GYYc[5][5] = {
    {-0.015625f, -0.0625f, -0.09375f, -0.0625f, -0.015625f},
    { 0.f,        0.f,      0.f,       0.f,      0.f},
    { 0.03125f,   0.125f,   0.1875f,   0.125f,   0.03125f},
    { 0.f,        0.f,      0.f,       0.f,      0.f},
    {-0.015625f, -0.0625f, -0.09375f, -0.0625f, -0.015625f}};

__global__ __launch_bounds__(256) void k_hessian(
    const float* __restrict__ lv, float* __restrict__ resp, int H, int W, int L)
{
    __shared__ __align__(16) float s[20 * 20];
    const int tx = threadIdx.x & 15, ty = threadIdx.x >> 4;
    const int bx = blockIdx.x * 16, by = blockIdx.y * 16;
    const float* im = lv + (size_t)blockIdx.z * H * W;
#if defined(__gfx1250__) && __has_builtin(__builtin_amdgcn_global_load_async_to_lds_b32)
    for (int i = threadIdx.x; i < 400; i += 256) {
        int r = i / 20, c = i - 20 * r;
        int gr = iclamp(by + r - 2, 0, H - 1);
        int gc = iclamp(bx + c - 2, 0, W - 1);
        __builtin_amdgcn_global_load_async_to_lds_b32(
            (__attribute__((address_space(1))) int*)(float*)
                &im[(size_t)gr * W + gc],
            (__attribute__((address_space(3))) int*)&s[i], 0, 0);
    }
  #if __has_builtin(__builtin_amdgcn_s_wait_asynccnt)
    __builtin_amdgcn_s_wait_asynccnt(0);
  #else
    asm volatile("s_wait_asynccnt 0" ::: "memory");
  #endif
#else
    for (int i = threadIdx.x; i < 400; i += 256) {
        int r = i / 20, c = i - 20 * r;
        int gr = iclamp(by + r - 2, 0, H - 1);
        int gc = iclamp(bx + c - 2, 0, W - 1);
        s[i] = im[(size_t)gr * W + gc];
    }
#endif
    __syncthreads();
    const int x = bx + tx, y = by + ty;
    if (x < W && y < H) {
        float dxx = 0.f, dxy = 0.f, dyy = 0.f;
        #pragma unroll
        for (int i = 0; i < 5; ++i)
            #pragma unroll
            for (int j = 0; j < 5; ++j) {
                float v = s[(ty + i) * 20 + tx + j];
                dxx += v * GXXc[i][j];
                dxy += v * GXYc[i][j];
                dyy += v * GYYc[i][j];
            }
        int l = blockIdx.z % L;
        float sg = 1.6f * __powf(1.25992105f, (float)l);
        float sig4 = (sg * sg) * (sg * sg);
        resp[(size_t)blockIdx.z * H * W + (size_t)y * W + x] =
            (dxx * dyy - dxy * dxy) * sig4;
    }
}

// ---------------------------------------------------------------------------
// Soft-argmax candidate (3x3x3 window over 5-level response volume)
// ---------------------------------------------------------------------------
struct Cand { float resp, sm, xw, yh; };

__device__ __forceinline__ Cand compute_cand(
    const float* __restrict__ resp, int Lfull, int b, int d, int y, int x,
    int H, int W, float smul, float xmul, float ymul)
{
    const int D = 5;
    const size_t plane = (size_t)H * W;
    const float* base = resp + (size_t)b * Lfull * plane;
    float m = -3.0e38f;
    for (int dd = d - 1; dd <= d + 1; ++dd) {
        if (dd < 0 || dd >= D) continue;
        for (int yy = y - 1; yy <= y + 1; ++yy) {
            if (yy < 0 || yy >= H) continue;
            for (int xx = x - 1; xx <= x + 1; ++xx) {
                if (xx < 0 || xx >= W) continue;
                float v = base[dd * plane + (size_t)yy * W + xx];
                m = fmaxf(m, v);
            }
        }
    }
    float den = 0.f, cd = 0.f, cx = 0.f, cy = 0.f, vs = 0.f;
    for (int dd = d - 1; dd <= d + 1; ++dd) {
        if (dd < 0 || dd >= D) continue;
        for (int yy = y - 1; yy <= y + 1; ++yy) {
            if (yy < 0 || yy >= H) continue;
            for (int xx = x - 1; xx <= x + 1; ++xx) {
                if (xx < 0 || xx >= W) continue;
                float v = base[dd * plane + (size_t)yy * W + xx];
                float e = __expf(v - m);
                den += e; vs += e * v;
                cd += e * (float)(dd - d);
                cx += e * (float)(xx - x);
                cy += e * (float)(yy - y);
            }
        }
    }
    den += 1e-8f;
    float fd = (float)d + cd / den;
    float fx = (float)x + cx / den;
    float fy = (float)y + cy / den;
    float val = vs / den;
    float scale = 1.6f * exp2f(fd * (1.f / 3.f));
    float s = 6.f * scale;
    bool good = (fx - s > 0.f) && (fx + s < (float)(W - 1)) &&
                (fy - s > 0.f) && (fy + s < (float)(H - 1));
    Cand c;
    c.resp = good ? val : 0.f;
    c.sm = s * smul;
    c.xw = fx * xmul;
    c.yh = fy * ymul;
    return c;
}

__device__ __forceinline__ unsigned bin_of(float f)
{
    unsigned u = __float_as_uint(f);
    u = (u & 0x80000000u) ? ~u : (u | 0x80000000u);  // order-preserving map
    return u >> 22;                                   // 1024 bins
}

__global__ void k_init(unsigned* hist, unsigned* cnts, int B)
{
    int n = B * 1024 + B * 2;
    for (int i = threadIdx.x + blockIdx.x * blockDim.x; i < n;
         i += blockDim.x * gridDim.x) {
        if (i < B * 1024) hist[i] = 0u;
        else              cnts[i - B * 1024] = 0u;
    }
}

__global__ void k_hist(const float* __restrict__ resp, int Lfull,
                       int H, int W, int B, unsigned* hist)
{
    long long N = (long long)B * 5 * H * W;
    long long idx = (long long)blockIdx.x * blockDim.x + threadIdx.x;
    if (idx >= N) return;
    int x = (int)(idx % W); long long t = idx / W;
    int y = (int)(t % H);   t /= H;
    int d = (int)(t % 5);   int b = (int)(t / 5);
    Cand c = compute_cand(resp, Lfull, b, d, y, x, H, W, 1.f, 1.f, 1.f);
    atomicAdd(&hist[b * 1024 + bin_of(c.resp)], 1u);
}

__global__ void k_select(const unsigned* __restrict__ hist, int B, int* thresh)
{
    int b = threadIdx.x;
    if (b < B) {
        unsigned cum = 0; int T = 0;
        for (int bin = 1023; bin >= 0; --bin) {
            cum += hist[b * 1024 + bin];
            if (cum >= 2000u) { T = bin; break; }
        }
        thresh[b] = T;
    }
}

__global__ void k_compact(const float* __restrict__ resp, int Lfull,
                          int H, int W, int B, const int* __restrict__ thresh,
                          unsigned* cnts, float* cand,
                          float smul, float xmul, float ymul)
{
    long long N = (long long)B * 5 * H * W;
    long long idx = (long long)blockIdx.x * blockDim.x + threadIdx.x;
    if (idx >= N) return;
    int x = (int)(idx % W); long long t = idx / W;
    int y = (int)(t % H);   t /= H;
    int d = (int)(t % 5);   int b = (int)(t / 5);
    Cand c = compute_cand(resp, Lfull, b, d, y, x, H, W, smul, xmul, ymul);
    int bn = (int)bin_of(c.resp);
    int T = thresh[b];
    if (bn > T) {
        unsigned slot = atomicAdd(&cnts[b * 2 + 0], 1u);
        if (slot < 2048u) {
            float* pp = cand + ((size_t)b * 4096 + slot) * 4;
            pp[0] = c.resp; pp[1] = c.sm; pp[2] = c.xw; pp[3] = c.yh;
        }
    } else if (bn == T) {
        unsigned slot = atomicAdd(&cnts[b * 2 + 1], 1u);
        if (slot < 2048u) {
            float* pp = cand + ((size_t)b * 4096 + 2048 + slot) * 4;
            pp[0] = c.resp; pp[1] = c.sm; pp[2] = c.xw; pp[3] = c.yh;
        }
    }
}

// Bitonic sort 4096 entries per batch, write top-2000 LAFs + responses
__global__ __launch_bounds__(1024) void k_sort(
    const float* __restrict__ cand, const unsigned* __restrict__ cnts,
    float* __restrict__ out, int B)
{
    __shared__ float key[4096];
    __shared__ int   sid[4096];
    const int b = blockIdx.x;
    unsigned ca = cnts[b * 2], cb2 = cnts[b * 2 + 1];
    if (ca > 2048u) ca = 2048u;
    if (cb2 > 2048u) cb2 = 2048u;
    for (int i = threadIdx.x; i < 4096; i += 1024) {
        bool valid = (i < 2048) ? (i < (int)ca) : ((i - 2048) < (int)cb2);
        key[i] = valid ? cand[((size_t)b * 4096 + i) * 4] : -3.0e38f;
        sid[i] = i;
    }
    __syncthreads();
    for (int k = 2; k <= 4096; k <<= 1) {
        for (int j = k >> 1; j > 0; j >>= 1) {
            for (int i = threadIdx.x; i < 4096; i += 1024) {
                int ixj = i ^ j;
                if (ixj > i) {
                    bool up = ((i & k) == 0);
                    float a = key[i], bb = key[ixj];
                    bool sw = up ? (a < bb) : (a > bb);   // descending
                    if (sw) {
                        key[i] = bb; key[ixj] = a;
                        int tt = sid[i]; sid[i] = sid[ixj]; sid[ixj] = tt;
                    }
                }
            }
            __syncthreads();
        }
    }
    float* lafs  = out;                              // (B,2000,2,3)
    float* resps = out + (size_t)B * 2000 * 6;       // (B,2000)
    for (int i = threadIdx.x; i < 2000; i += 1024) {
        float r = key[i];
        float sm = 0.f, xw = 0.f, yh = 0.f;
        if (r > -1.0e38f) {
            const float* pp = cand + ((size_t)b * 4096 + sid[i]) * 4;
            sm = pp[1]; xw = pp[2]; yh = pp[3];
        } else {
            r = 0.f;
        }
        float* Lp = lafs + ((size_t)b * 2000 + i) * 6;
        Lp[0] = sm; Lp[1] = 0.f; Lp[2] = xw;
        Lp[3] = 0.f; Lp[4] = sm; Lp[5] = yh;
        resps[(size_t)b * 2000 + i] = r;
    }
}

// cur [b][H][W] -> levels plane (b*L+0)
__global__ void k_copy_level(const float* __restrict__ src,
                             float* __restrict__ levels, int H, int W, int L)
{
    long long plane = (long long)H * W;
    long long n = 2 * plane;
    for (long long i = (long long)blockIdx.x * blockDim.x + threadIdx.x; i < n;
         i += (long long)blockDim.x * gridDim.x) {
        long long b = i / plane, rem = i - b * plane;
        levels[(size_t)b * L * plane + rem] = src[i];
    }
}

// levels plane (b*L+3) strided-2 -> dst [b][H2][W2]
__global__ void k_down(const float* __restrict__ levels,
                       float* __restrict__ dst, int H, int W, int L,
                       int H2, int W2)
{
    int x = blockIdx.x * blockDim.x + threadIdx.x;
    int y = blockIdx.y * blockDim.y + threadIdx.y;
    int b = blockIdx.z;
    if (x < W2 && y < H2) {
        dst[((size_t)b * H2 + y) * W2 + x] =
            levels[(((size_t)b * L + 3) * H + 2 * y) * W + 2 * x];
    }
}

// ---------------------------------------------------------------------------
extern "C" void kernel_launch(void* const* d_in, const int* in_sizes, int n_in,
                              void* d_out, int out_size, void* d_ws,
                              size_t ws_size, hipStream_t stream)
{
    (void)in_sizes; (void)n_in; (void)out_size;
    const float* img = (const float*)d_in[0];
    float* out = (float*)d_out;
    char* ws = (char*)d_ws;

    const int B = 2, Himg = 1024, Wimg = 1024, L = 6;

    // octave geometry + response-buffer offsets
    int oh[12], ow[12]; size_t roff[12]; int nOct = 0; size_t rtot = 0;
    {
        int Hc = Himg, Wc = Wimg;
        for (;;) {
            oh[nOct] = Hc; ow[nOct] = Wc; roff[nOct] = rtot;
            rtot += (size_t)B * L * Hc * Wc;
            nOct++;
            int nh = Hc / 2, nw = Wc / 2;
            if (nh <= 15 || nw <= 15) break;
            Hc = nh; Wc = nw;
        }
    }

    // workspace carve-out
    size_t off = 0;
    auto carve = [&](size_t bytes) {
        size_t o = off;
        off = (off + bytes + 255) & ~(size_t)255;
        return o;
    };
    unsigned* hist  = (unsigned*)(ws + carve((size_t)B * 1024 * 4));
    unsigned* cnts  = (unsigned*)(ws + carve((size_t)B * 2 * 4));
    int*      thr   = (int*)     (ws + carve((size_t)B * 4));
    float*    cand  = (float*)   (ws + carve((size_t)B * 4096 * 4 * 4));
    float*    curA  = (float*)   (ws + carve((size_t)B * Himg * Wimg * 4));
    float*    curB  = (float*)   (ws + carve((size_t)B * Himg * Wimg * 4));
    float*    levels= (float*)   (ws + carve((size_t)B * L * Himg * Wimg * 4));
    float*    rbase = (float*)   (ws + carve(rtot * 4));
    if (off > ws_size) return;  // workspace too small; nothing safe to do

    const double stepd = pow(2.0, 1.0 / 3.0);
    const float  kfac  = (float)sqrt(stepd * stepd - 1.0);

    k_init<<<8, 256, 0, stream>>>(hist, cnts, B);

    // initial blur of the input image -> curA
    {
        float sigma0 = sqrtf(1.6f * 1.6f - 0.25f);
        dim3 g((Wimg + BTILE - 1) / BTILE, (Himg + BTILE - 1) / BTILE, B);
        k_blur<<<g, 128, 0, stream>>>(img, curA, Himg, Wimg,
                                      (long long)Himg * Wimg,
                                      (long long)Himg * Wimg, sigma0);
    }

    for (int o = 0; o < nOct; ++o) {
        const int H = oh[o], W = ow[o];
        const long long plane = (long long)H * W;
        float* resp = rbase + roff[o];

        // levels[0] = cur
        k_copy_level<<<512, 256, 0, stream>>>(curA, levels, H, W, L);

        // blur chain levels[1..5]
        for (int l = 1; l < L; ++l) {
            float sigPrev = 1.6f * (float)pow(stepd, (double)(l - 1));
            float sb = sigPrev * kfac;
            dim3 g((W + BTILE - 1) / BTILE, (H + BTILE - 1) / BTILE, B);
            k_blur<<<g, 128, 0, stream>>>(
                levels + (size_t)(l - 1) * plane,
                levels + (size_t)l * plane,
                H, W, (long long)L * plane, (long long)L * plane, sb);
        }

        // Hessian response for all B*L planes
        {
            dim3 g((W + 15) / 16, (H + 15) / 16, B * L);
            k_hessian<<<g, 256, 0, stream>>>(levels, resp, H, W, L);
        }

        // histogram of soft-argmax responses
        {
            long long N = (long long)B * 5 * H * W;
            int blocks = (int)((N + 255) / 256);
            k_hist<<<blocks, 256, 0, stream>>>(resp, L, H, W, B, hist);
        }

        // downsample levels[3] for next octave
        if (o + 1 < nOct) {
            int H2 = oh[o + 1], W2 = ow[o + 1];
            dim3 bl(16, 16, 1);
            dim3 g((W2 + 15) / 16, (H2 + 15) / 16, B);
            k_down<<<g, bl, 0, stream>>>(levels, curB, H, W, L, H2, W2);
            float* t = curA; curA = curB; curB = t;
        }
    }

    // global threshold per batch, then compaction per octave
    k_select<<<1, 64, 0, stream>>>(hist, B, thr);
    for (int o = 0; o < nOct; ++o) {
        const int H = oh[o], W = ow[o];
        float* resp = rbase + roff[o];
        float mn = (float)(H < W ? H : W);
        float smul = (float)(Himg < Wimg ? Himg : Wimg) / mn;
        float xmul = (float)Wimg / (float)W;
        float ymul = (float)Himg / (float)H;
        long long N = (long long)B * 5 * H * W;
        int blocks = (int)((N + 255) / 256);
        k_compact<<<blocks, 256, 0, stream>>>(resp, L, H, W, B, thr, cnts,
                                              cand, smul, xmul, ymul);
    }

    // sort + emit (lafs then responses, flat)
    k_sort<<<B, 1024, 0, stream>>>(cand, cnts, out, B);
}
